// DCC_GNN_cosine_55121610277289
// MI455X (gfx1250) — matrix operand order repaired and checked by reading, not compiled, not run on hardware
//
#include <hip/hip_runtime.h>
#include <hip/hip_bf16.h>

typedef __attribute__((ext_vector_type(16))) _Float16 v16h;
typedef __attribute__((ext_vector_type(8)))  float    v8f;

// ---------------------------------------------------------------------------
// order-preserving float <-> uint encoding for atomicMax-based segment max
// ---------------------------------------------------------------------------
__device__ __forceinline__ unsigned encf(float f) {
    unsigned b = __float_as_uint(f);
    return (b & 0x80000000u) ? ~b : (b | 0x80000000u);
}
__device__ __forceinline__ float decf(unsigned k) {
    unsigned b = (k & 0x80000000u) ? (k & 0x7fffffffu) : ~k;
    return __uint_as_float(b);
}

// ---------------------------------------------------------------------------
// Weight transpose + cast: Wt[n][k] = (f16) W[k][n]
// ---------------------------------------------------------------------------
__global__ void wtc_kernel(const float* __restrict__ W, _Float16* __restrict__ Wt,
                           int K, int N) {
    int t = blockIdx.x * blockDim.x + threadIdx.x;
    if (t >= K * N) return;
    int n = t / K, k = t - n * K;
    Wt[t] = (_Float16)W[(size_t)k * N + n];
}

// ---------------------------------------------------------------------------
// Embedding gather: h[n][c] = emb[tokens[n]][c] ; also cast to f16
// ---------------------------------------------------------------------------
__global__ void embed_kernel(const int* __restrict__ tokens,
                             const float* __restrict__ emb,
                             float* __restrict__ h, _Float16* __restrict__ hf,
                             int NT, int IN) {
    int t = blockIdx.x * blockDim.x + threadIdx.x;
    if (t >= NT * IN) return;
    int n = t / IN, c = t - n * IN;
    float v = emb[(size_t)tokens[n] * IN + c];
    h[t] = v;
    hf[t] = (_Float16)v;
}

// ---------------------------------------------------------------------------
// WMMA GEMM: C[M,N] = A[M,K](f16, row-major) x Bt[N,K](f16, W transposed)
// block = 256 threads = 8 waves; block tile = 16(M) x 128(N); wave = 16x16
// ---------------------------------------------------------------------------
__global__ __launch_bounds__(256) void gemm16_wmma(
    const _Float16* __restrict__ A, const _Float16* __restrict__ Bt,
    float* __restrict__ C, int K, int N)
{
    __shared__ __align__(16) _Float16 lA[16 * 256];   // 16 rows x K (K<=256)
    const int tid  = threadIdx.x;
    const int wave = tid >> 5;
    const int lane = tid & 31;
    const int m0 = blockIdx.x << 4;
    const int n0 = (blockIdx.y << 7) + (wave << 4);

    // Rows m0..m0+15 of row-major A are one contiguous 16*K block: flat copy.
    const uint4* gsrc = (const uint4*)(A + (size_t)m0 * K);
    uint4* ldst = (uint4*)lA;
    const int chunks = (16 * K) >> 3;                 // 16B chunks
    for (int i = tid; i < chunks; i += 256) ldst[i] = gsrc[i];
    __syncthreads();

    const int row = lane & 15;
    const int hi  = lane >> 4;                        // 0: low K half, 1: high
    const _Float16* ap = lA + row * K + hi * 8;       // A frag: K {0-7,16-23}/{8-15,24-31}
    const _Float16* bp = Bt + (size_t)(n0 + row) * K + hi * 8;  // B frag: K {0-7,16-23}/{8-15,24-31}

    v8f acc = {0.f, 0.f, 0.f, 0.f, 0.f, 0.f, 0.f, 0.f};
    for (int kk = 0; kk < K; kk += 32) {
        union { v16h v; float4 f[2]; } a, b;
        a.f[0] = *(const float4*)(ap + kk);
        a.f[1] = *(const float4*)(ap + kk + 16);
        b.f[0] = *(const float4*)(bp + kk);
        b.f[1] = *(const float4*)(bp + kk + 16);
        acc = __builtin_amdgcn_wmma_f32_16x16x32_f16(
            /*neg_a=*/false, a.v, /*neg_b=*/false, b.v,
            /*c_mod=*/(short)0, acc, /*reuse_a=*/false, /*reuse_b=*/false);
    }

    // C layout: VGPR r -> row m0 + r + 8*hi, col n0 + (lane&15)
    float* cp = C + (size_t)(m0 + hi * 8) * N + n0 + row;
    #pragma unroll
    for (int r = 0; r < 8; ++r) cp[(size_t)r * N] = acc[r];
}

// ---------------------------------------------------------------------------
// el[n,h] = <z[n,h,:], al[h,:]> ; er likewise
// ---------------------------------------------------------------------------
__global__ void attn_lr_kernel(const float* __restrict__ z,
                               const float* __restrict__ al,
                               const float* __restrict__ ar,
                               float* __restrict__ el, float* __restrict__ er,
                               int NT, int H, int DH) {
    int t = blockIdx.x * blockDim.x + threadIdx.x;
    if (t >= NT * H) return;
    int n = t / H, h = t - n * H;
    const float* zp  = z + (size_t)n * H * DH + h * DH;
    const float* alp = al + h * DH;
    const float* arp = ar + h * DH;
    float sl = 0.f, sr = 0.f;
    for (int d = 0; d < DH; ++d) { float v = zp[d]; sl += v * alp[d]; sr += v * arp[d]; }
    el[t] = sl; er[t] = sr;
}

// ---------------------------------------------------------------------------
// Edge scores: e = leaky_relu(el[src]+er[dst]); segment max via atomicMax(enc)
// ---------------------------------------------------------------------------
__global__ void edge_score_kernel(const int* __restrict__ src, const int* __restrict__ dst,
                                  const float* __restrict__ el, const float* __restrict__ er,
                                  float* __restrict__ esc, unsigned* __restrict__ mkey,
                                  int E, int H) {
    int t = blockIdx.x * blockDim.x + threadIdx.x;
    if (t >= E * H) return;
    int e = t / H, h = t - e * H;
    float x = el[src[e] * H + h] + er[dst[e] * H + h];
    x = (x > 0.f) ? x : 0.2f * x;
    esc[t] = x;
    atomicMax(&mkey[dst[e] * H + h], encf(x));
}

// ---------------------------------------------------------------------------
// ex = exp(e - m[dst]); segment sum via atomicAdd
// ---------------------------------------------------------------------------
__global__ void edge_exp_kernel(const int* __restrict__ dst,
                                float* __restrict__ esc,
                                const unsigned* __restrict__ mkey,
                                float* __restrict__ ssum, int E, int H) {
    int t = blockIdx.x * blockDim.x + threadIdx.x;
    if (t >= E * H) return;
    int e = t / H, h = t - e * H;
    int d = dst[e];
    unsigned k = mkey[d * H + h];
    float m = (k == 0u) ? 0.f : decf(k);     // empty segment -> m = 0 (ref semantics)
    float ex = __expf(esc[t] - m);
    esc[t] = ex;
    atomicAdd(&ssum[d * H + h], ex);
}

// ---------------------------------------------------------------------------
// Aggregation: agg[dst] += alpha * z[src]   (one wave per edge, lanes = feats)
// ---------------------------------------------------------------------------
__global__ __launch_bounds__(256) void edge_aggr_kernel(
    const int* __restrict__ src, const int* __restrict__ dst,
    const float* __restrict__ esc, const float* __restrict__ ssum,
    const float* __restrict__ z, float* __restrict__ agg,
    int E, int H, int DH) {
    int w = (blockIdx.x * blockDim.x + threadIdx.x) >> 5;
    int lane = threadIdx.x & 31;
    if (w >= E) return;
    int s = src[w], d = dst[w];
    int F = H * DH;
    const float* zp = z + (size_t)s * F;
    float* op = agg + (size_t)d * F;
    for (int h = 0; h < H; ++h) {
        float alpha = esc[(size_t)w * H + h] / (ssum[d * H + h] + 1e-9f);
        for (int dd = lane; dd < DH; dd += 32)
            atomicAdd(&op[h * DH + dd], alpha * zp[h * DH + dd]);
    }
}

// ---------------------------------------------------------------------------
// ELU + cast to f16 for next layer's GEMM
// ---------------------------------------------------------------------------
__global__ void elu_cast_kernel(const float* __restrict__ agg,
                                float* __restrict__ h, _Float16* __restrict__ hf,
                                int n) {
    int t = blockIdx.x * blockDim.x + threadIdx.x;
    if (t >= n) return;
    float v = agg[t];
    float y = (v > 0.f) ? v : (__expf(v) - 1.f);
    h[t] = y;
    hf[t] = (_Float16)y;
}

// ---------------------------------------------------------------------------
// Final layer: mean over heads -> hfin[NT, GD]
// ---------------------------------------------------------------------------
__global__ void head_mean_kernel(const float* __restrict__ agg,
                                 float* __restrict__ hfin,
                                 int NT, int H, int GD) {
    int t = blockIdx.x * blockDim.x + threadIdx.x;
    if (t >= NT * GD) return;
    int n = t / GD, g = t - n * GD;
    float s = 0.f;
    for (int h = 0; h < H; ++h) s += agg[(size_t)n * H * GD + h * GD + g];
    hfin[t] = s * (1.0f / H);
}

// ---------------------------------------------------------------------------
// Readout: per-graph node mean -> v[B, GD]
// ---------------------------------------------------------------------------
__global__ void graph_mean_kernel(const float* __restrict__ hfin,
                                  float* __restrict__ v, int Nn, int B, int GD) {
    int t = blockIdx.x * blockDim.x + threadIdx.x;
    if (t >= B * GD) return;
    int b = t / GD, g = t - b * GD;
    const float* p = hfin + (size_t)b * Nn * GD + g;
    float s = 0.f;
    for (int i = 0; i < Nn; ++i) s += p[(size_t)i * GD];
    v[t] = s * (1.0f / Nn);
}

// ---------------------------------------------------------------------------
// Cosine similarity
// ---------------------------------------------------------------------------
__global__ void cosine_kernel(const float* __restrict__ v1, const float* __restrict__ v2,
                              float* __restrict__ out, int B, int GD) {
    int b = blockIdx.x * blockDim.x + threadIdx.x;
    if (b >= B) return;
    float num = 0.f, a = 0.f, c = 0.f;
    for (int g = 0; g < GD; ++g) {
        float x = v1[b * GD + g], y = v2[b * GD + g];
        num += x * y; a += x * x; c += y * y;
    }
    out[b] = num / (fmaxf(sqrtf(a), 1e-8f) * fmaxf(sqrtf(c), 1e-8f));
}

// ---------------------------------------------------------------------------
extern "C" void kernel_launch(void* const* d_in, const int* in_sizes, int n_in,
                              void* d_out, int out_size, void* d_ws, size_t ws_size,
                              hipStream_t stream)
{
    const int NT = 32000, E = 512000, B = 32, Nn = 1000;
    const int IN = 128, H = 4, HID = 256, GD = 128;

    const int* tok[2] = {(const int*)d_in[0], (const int*)d_in[3]};
    const int* src[2] = {(const int*)d_in[1], (const int*)d_in[4]};
    const int* dst[2] = {(const int*)d_in[2], (const int*)d_in[5]};
    const float* emb = (const float*)d_in[6];
    const float* W[4]  = {(const float*)d_in[7],  (const float*)d_in[10],
                          (const float*)d_in[13], (const float*)d_in[16]};
    const float* al[4] = {(const float*)d_in[8],  (const float*)d_in[11],
                          (const float*)d_in[14], (const float*)d_in[17]};
    const float* ar[4] = {(const float*)d_in[9],  (const float*)d_in[12],
                          (const float*)d_in[15], (const float*)d_in[18]};
    const int Kd[4]  = {IN, HID, HID, HID};
    const int Nd[4]  = {HID, HID, HID, H * GD};
    const int DHd[4] = {64, 64, 64, GD};

    // bump allocator on workspace (256B aligned)
    char* ws = (char*)d_ws;
    size_t off = 0;
    auto alloc = [&](size_t bytes) -> char* {
        char* p = ws + off;
        off += (bytes + 255) & ~(size_t)255;
        return p;
    };

    float*    hf32 = (float*)   alloc((size_t)NT * HID * 4);
    _Float16* hf16 = (_Float16*)alloc((size_t)NT * HID * 2);
    float*    z    = (float*)   alloc((size_t)NT * 512 * 4);
    float*    agg  = (float*)   alloc((size_t)NT * 512 * 4);
    float*    el   = (float*)   alloc((size_t)NT * H * 4);
    float*    er   = (float*)   alloc((size_t)NT * H * 4);
    unsigned* mkey = (unsigned*)alloc((size_t)NT * H * 4);
    float*    ssum = (float*)   alloc((size_t)NT * H * 4);
    float*    esc  = (float*)   alloc((size_t)E * H * 4);
    _Float16* Wt[4];
    for (int l = 0; l < 4; ++l) Wt[l] = (_Float16*)alloc((size_t)Kd[l] * Nd[l] * 2);
    float* v[2];
    v[0] = (float*)alloc((size_t)B * GD * 4);
    v[1] = (float*)alloc((size_t)B * GD * 4);
    float* hfin = z;   // z is dead by the time head_mean runs; reuse

    // Pre-transpose + cast weights to f16 [N][K] once
    for (int l = 0; l < 4; ++l) {
        int tot = Kd[l] * Nd[l];
        wtc_kernel<<<(tot + 255) / 256, 256, 0, stream>>>(W[l], Wt[l], Kd[l], Nd[l]);
    }

    for (int g = 0; g < 2; ++g) {
        embed_kernel<<<((NT * IN) + 255) / 256, 256, 0, stream>>>(
            tok[g], emb, hf32, hf16, NT, IN);
        int Fin = IN;
        for (int l = 0; l < 4; ++l) {
            const int Fout = Nd[l], DH = DHd[l];

            dim3 grid(NT / 16, Fout / 128);
            gemm16_wmma<<<grid, 256, 0, stream>>>(hf16, Wt[l], z, Fin, Fout);

            attn_lr_kernel<<<((NT * H) + 255) / 256, 256, 0, stream>>>(
                z, al[l], ar[l], el, er, NT, H, DH);

            hipMemsetAsync(mkey, 0, (size_t)NT * H * 4, stream);
            hipMemsetAsync(ssum, 0, (size_t)NT * H * 4, stream);
            hipMemsetAsync(agg,  0, (size_t)NT * Fout * 4, stream);

            edge_score_kernel<<<((E * H) + 255) / 256, 256, 0, stream>>>(
                src[g], dst[g], el, er, esc, mkey, E, H);
            edge_exp_kernel<<<((E * H) + 255) / 256, 256, 0, stream>>>(
                dst[g], esc, mkey, ssum, E, H);
            edge_aggr_kernel<<<(E + 7) / 8, 256, 0, stream>>>(
                src[g], dst[g], esc, ssum, z, agg, E, H, DH);

            if (l < 3) {
                elu_cast_kernel<<<((NT * Fout) + 255) / 256, 256, 0, stream>>>(
                    agg, hf32, hf16, NT * Fout);
                Fin = Fout;
            } else {
                head_mean_kernel<<<((NT * GD) + 255) / 256, 256, 0, stream>>>(
                    agg, hfin, NT, H, GD);
            }
        }
        graph_mean_kernel<<<((B * GD) + 255) / 256, 256, 0, stream>>>(
            hfin, v[g], Nn, B, GD);
    }

    cosine_kernel<<<1, 64, 0, stream>>>(v[0], v[1], (float*)d_out, B, GD);
}
